// Attention_59450937311370
// MI455X (gfx1250) — compile-verified, matrix operand order checked by reading, not run on hardware
//
#include <hip/hip_runtime.h>

typedef __attribute__((ext_vector_type(16))) __bf16 v16bf;
typedef __attribute__((ext_vector_type(8)))  __bf16 bf16x8;
typedef __attribute__((ext_vector_type(8)))  float  v8f;
typedef __attribute__((ext_vector_type(4)))  unsigned int u32x4;
typedef __attribute__((ext_vector_type(8)))  int i32x8;
typedef __attribute__((ext_vector_type(4)))  int i32x4;

#define DIM 512
#define HEADS 4
#define HEAD_DIM 128
#define SEQ 4096
#define BATCH 2
#define ROWS (BATCH * SEQ)          // 8192
#define SCALE 0.08838834764831845f  // 128^-0.5
#define LOG2E 1.4426950408889634f

// Build a 16-element bf16 A/B WMMA fragment from two contiguous 8-element runs.
__device__ inline v16bf load_frag(const __bf16* p0, const __bf16* p1) {
    union { v16bf v; bf16x8 h[2]; } u;
    u.h[0] = *(const bf16x8*)p0;
    u.h[1] = *(const bf16x8*)p1;
    return u.v;
}

__device__ inline v8f wmma_bf16(v16bf a, v16bf b, v8f c) {
    return __builtin_amdgcn_wmma_f32_16x16x32_bf16(false, a, false, b, (short)0, c, false, false);
}

// ---- Tensor Data Mover: 2D bf16 tile (tile_d1 rows x tile_d0 elems) -> LDS ----
// D# packing per CDNA5 ISA.
// clang-23 toolchain: 6-arg builtin (g0, g1, g2, g3, g_extra, cpol).
__device__ inline void tdm_load_2d_bf16(unsigned lds_off, const void* gaddr,
                                        unsigned tensor_d0, unsigned tensor_d1,
                                        unsigned long long stride0,
                                        unsigned tile_d0, unsigned tile_d1) {
    unsigned long long ga = (unsigned long long)gaddr;
    u32x4 g0;
    g0[0] = 1u;                                              // count=1, user descriptor
    g0[1] = lds_off;                                         // lds_addr
    g0[2] = (unsigned)ga;                                    // global_addr[31:0]
    g0[3] = (unsigned)((ga >> 32) & 0x1FFFFFFu) | (2u << 30); // global_addr[56:32] | type=2
    i32x8 g1;
    g1[0] = (int)(1u << 16);                                 // wg_mask=0, data_size=1 (2 bytes)
    g1[1] = (int)((tensor_d0 & 0xFFFFu) << 16);              // tensor_dim0[15:0]
    g1[2] = (int)((tensor_d0 >> 16) | ((tensor_d1 & 0xFFFFu) << 16));
    g1[3] = (int)((tensor_d1 >> 16) | ((tile_d0 & 0xFFFFu) << 16));
    g1[4] = (int)(tile_d1 & 0xFFFFu);                        // tile_dim1, tile_dim2=0
    g1[5] = (int)(unsigned)(stride0 & 0xFFFFFFFFull);        // tensor_dim0_stride[31:0]
    g1[6] = (int)(unsigned)((stride0 >> 32) & 0xFFFFu);      // stride[47:32], dim1_stride=0
    g1[7] = 0;
    i32x4 z4 = {0, 0, 0, 0};
    i32x8 z8 = {0, 0, 0, 0, 0, 0, 0, 0};
    __builtin_amdgcn_tensor_load_to_lds(g0, g1, z4, z4, z8, 0);
}

__device__ inline unsigned lds_off_of(const void* p) {
    // Flat shared addresses map to LDS via addr[31:0] (ISA 10.2 aperture table).
    return (unsigned)(unsigned long long)p;
}

// ---------------- LayerNorm: f32 in -> bf16 out --------------------------
__global__ __launch_bounds__(256) void ln_kernel(const float* __restrict__ x,
                                                 const float* __restrict__ gamma,
                                                 const float* __restrict__ beta,
                                                 __bf16* __restrict__ xn) {
    __shared__ float red[256];
    const int row = blockIdx.x;
    const int t = threadIdx.x;
    const float* xr = x + (size_t)row * DIM;
    float v0 = xr[t], v1 = xr[t + 256];
    red[t] = v0 + v1;
    __syncthreads();
    for (int off = 128; off > 0; off >>= 1) {
        if (t < off) red[t] += red[t + off];
        __syncthreads();
    }
    float mu = red[0] * (1.0f / DIM);
    __syncthreads();
    float d0 = v0 - mu, d1 = v1 - mu;
    red[t] = d0 * d0 + d1 * d1;
    __syncthreads();
    for (int off = 128; off > 0; off >>= 1) {
        if (t < off) red[t] += red[t + off];
        __syncthreads();
    }
    float rstd = rsqrtf(red[0] * (1.0f / DIM) + 1e-5f);
    __bf16* o = xn + (size_t)row * DIM;
    o[t]       = (__bf16)(d0 * rstd * gamma[t] + beta[t]);
    o[t + 256] = (__bf16)(d1 * rstd * gamma[t + 256] + beta[t + 256]);
}

// --------------- Transpose + convert weights to bf16 ---------------------
__global__ void transpose_cvt(const float* __restrict__ w, __bf16* __restrict__ wt,
                              int K, int N) {
    int idx = blockIdx.x * blockDim.x + threadIdx.x;
    if (idx < K * N) {
        int k = idx / N, n = idx % N;
        wt[(size_t)n * K + k] = (__bf16)w[idx];
    }
}

// --------------- QKV GEMM: [8192,512] @ [512,1536] + bias ----------------
__global__ __launch_bounds__(256) void qkv_gemm(const __bf16* __restrict__ xn,
                                                const __bf16* __restrict__ wT,
                                                const float* __restrict__ bias,
                                                __bf16* __restrict__ q,
                                                __bf16* __restrict__ k,
                                                __bf16* __restrict__ vT) {
    const int NT = 96;  // 1536 / 16
    const int tile = blockIdx.x * 8 + (threadIdx.x >> 5);
    const int mt = tile / NT, nt = tile % NT;
    const int lane = threadIdx.x & 31;
    const int low = lane & 15;       // A row index / B column index
    const int hi = lane >> 4;
    const int base = hi * 8;

    const __bf16* arow = xn + (size_t)(mt * 16 + low) * DIM;
    const __bf16* brow = wT + (size_t)(nt * 16 + low) * DIM;

    v8f acc = {0.f, 0.f, 0.f, 0.f, 0.f, 0.f, 0.f, 0.f};
#pragma unroll 4
    for (int k0 = 0; k0 < DIM; k0 += 32) {
        v16bf a = load_frag(arow + k0 + base, arow + k0 + 16 + base);
        v16bf b = load_frag(brow + k0 + base, brow + k0 + 16 + base);
        acc = wmma_bf16(a, b, acc);
    }

    const int col = nt * 16 + low;        // output feature 0..1535
    const float bv = bias[col];
    const int sel = col >> 9;             // 0=Q 1=K 2=V
    const int head = (col >> 7) & 3;
    const int hd = col & 127;
#pragma unroll
    for (int r = 0; r < 8; r++) {
        int m = mt * 16 + r + hi * 8;     // global row (C layout)
        int bb = m >> 12, nn = m & 4095;
        __bf16 val = (__bf16)(acc[r] + bv);
        size_t bh = (size_t)(bb * HEADS + head);
        if (sel == 0)      q[(bh * SEQ + nn) * HEAD_DIM + hd] = val;
        else if (sel == 1) k[(bh * SEQ + nn) * HEAD_DIM + hd] = val;
        else               vT[(bh * HEAD_DIM + hd) * SEQ + nn] = val;
    }
}

// --------------- Flash attention: one (bh, 64-query tile) per block ------
// K/V tiles staged into LDS by the TDM (double buffered), shared by 4 waves.
__global__ __launch_bounds__(128) void attn_kernel(const __bf16* __restrict__ q,
                                                   const __bf16* __restrict__ kk,
                                                   const __bf16* __restrict__ vT,
                                                   __bf16* __restrict__ aout) {
    __shared__ __align__(16) __bf16 Kt[2][32][HEAD_DIM];   // 2 x 8 KB
    __shared__ __align__(16) __bf16 Vt[2][HEAD_DIM][32];   // 2 x 8 KB
    __shared__ __align__(16) __bf16 plds[4][16][32];       // per-wave P staging, 4 KB

    const int w = threadIdx.x >> 5;
    const int lane = threadIdx.x & 31;
    const int col = lane & 15;
    const int hi = lane >> 4;
    const int base = hi * 8;

    const int bh = blockIdx.x >> 6;   // 0..7
    const int qt = blockIdx.x & 63;
    const int qbase = qt * 64 + w * 16;

    const __bf16* Q = q  + (size_t)bh * SEQ * HEAD_DIM;
    const __bf16* K = kk + (size_t)bh * SEQ * HEAD_DIM;
    const __bf16* V = vT + (size_t)bh * HEAD_DIM * SEQ;

    // Preload the 16x128 query A-fragments (4 k-chunks of 32)
    v16bf aQ[4];
    const __bf16* qrow = Q + (size_t)(qbase + col) * HEAD_DIM;
#pragma unroll
    for (int c = 0; c < 4; c++)
        aQ[c] = load_frag(qrow + c * 32 + base, qrow + c * 32 + 16 + base);

    v8f o[8];
#pragma unroll
    for (int j = 0; j < 8; j++) o[j] = (v8f){0.f,0.f,0.f,0.f,0.f,0.f,0.f,0.f};
    float m[8], l[8];
#pragma unroll
    for (int r = 0; r < 8; r++) { m[r] = -3.0e38f; l[r] = 0.f; }

    // Prologue: TDM-fill buffer 0 with the first 32-key tile.
    if (w == 0) {
        tdm_load_2d_bf16(lds_off_of(&Kt[0][0][0]), K, HEAD_DIM, SEQ,
                         HEAD_DIM, HEAD_DIM, 32);
        tdm_load_2d_bf16(lds_off_of(&Vt[0][0][0]), V, SEQ, HEAD_DIM,
                         SEQ, 32, HEAD_DIM);
        __builtin_amdgcn_s_wait_tensorcnt(0);
    }
    __syncthreads();

    int cur = 0;
    for (int kt = 0; kt < SEQ; kt += 32) {
        const int nxt = cur ^ 1;
        // Prefetch next key tile into the other buffer while we compute.
        if (w == 0 && kt + 32 < SEQ) {
            tdm_load_2d_bf16(lds_off_of(&Kt[nxt][0][0]),
                             K + (size_t)(kt + 32) * HEAD_DIM,
                             HEAD_DIM, SEQ, HEAD_DIM, HEAD_DIM, 32);
            tdm_load_2d_bf16(lds_off_of(&Vt[nxt][0][0]),
                             V + (kt + 32),
                             SEQ, HEAD_DIM, SEQ, 32, HEAD_DIM);
        }

        // S = Q Kt for 32 keys -> two 16x16 C tiles (B fragments from LDS)
        v8f s0 = {0.f,0.f,0.f,0.f,0.f,0.f,0.f,0.f};
        v8f s1 = {0.f,0.f,0.f,0.f,0.f,0.f,0.f,0.f};
        const __bf16* krow0 = &Kt[cur][col][0];
        const __bf16* krow1 = &Kt[cur][16 + col][0];
#pragma unroll
        for (int c = 0; c < 4; c++) {
            v16bf b0 = load_frag(krow0 + c * 32 + base, krow0 + c * 32 + 16 + base);
            s0 = wmma_bf16(aQ[c], b0, s0);
            v16bf b1 = load_frag(krow1 + c * 32 + base, krow1 + c * 32 + 16 + base);
            s1 = wmma_bf16(aQ[c], b1, s1);
        }

        // Online softmax per row (rows live across 16-lane halves)
#pragma unroll
        for (int r = 0; r < 8; r++) {
            float a0 = s0[r] * SCALE, a1 = s1[r] * SCALE;
            float t = fmaxf(a0, a1);
#pragma unroll
            for (int msk = 8; msk >= 1; msk >>= 1) t = fmaxf(t, __shfl_xor(t, msk, 32));
            float mn = fmaxf(m[r], t);
            float alpha = exp2f((m[r] - mn) * LOG2E);
            float p0 = exp2f((a0 - mn) * LOG2E);
            float p1 = exp2f((a1 - mn) * LOG2E);
            float rs = p0 + p1;
#pragma unroll
            for (int msk = 8; msk >= 1; msk >>= 1) rs += __shfl_xor(rs, msk, 32);
            l[r] = l[r] * alpha + rs;
            m[r] = mn;
#pragma unroll
            for (int j = 0; j < 8; j++) o[j][r] *= alpha;
            plds[w][r + hi * 8][col]      = (__bf16)p0;
            plds[w][r + hi * 8][16 + col] = (__bf16)p1;
        }
        asm volatile("s_wait_dscnt 0x0" ::: "memory");

        // O += P @ V  (K-dim = 32 keys, 8 hd column tiles, B from LDS)
        v16bf aP = load_frag(&plds[w][col][base], &plds[w][col][16 + base]);
#pragma unroll
        for (int j = 0; j < 8; j++) {
            const __bf16* vrow = &Vt[cur][j * 16 + col][0];
            v16bf bV = load_frag(vrow + base, vrow + 16 + base);
            o[j] = wmma_bf16(aP, bV, o[j]);
        }

        // All LDS reads of buffer `cur` must land before TDM may refill it
        // (in the next iteration's prefetch); also wait for the prefetch to complete.
        asm volatile("s_wait_dscnt 0x0" ::: "memory");
        if (w == 0) __builtin_amdgcn_s_wait_tensorcnt(0);
        __syncthreads();
        cur = nxt;
    }

    // Normalize and emit bf16 [b, n, 512] for the output projection
    const int b = bh >> 2, h = bh & 3;
#pragma unroll
    for (int j = 0; j < 8; j++) {
#pragma unroll
        for (int r = 0; r < 8; r++) {
            int qi = qbase + r + hi * 8;
            float val = o[j][r] / l[r];
            aout[((size_t)b * SEQ + qi) * DIM + h * HEAD_DIM + j * 16 + col] = (__bf16)val;
        }
    }
}

// --------------- Output GEMM: [8192,512] @ [512,512] + bias -> f32 -------
__global__ __launch_bounds__(256) void out_gemm(const __bf16* __restrict__ aout,
                                                const __bf16* __restrict__ wT,
                                                const float* __restrict__ bias,
                                                float* __restrict__ out) {
    const int NT = 32;  // 512 / 16
    const int tile = blockIdx.x * 8 + (threadIdx.x >> 5);
    const int mt = tile / NT, nt = tile % NT;
    const int lane = threadIdx.x & 31;
    const int low = lane & 15;
    const int hi = lane >> 4;
    const int base = hi * 8;

    const __bf16* arow = aout + (size_t)(mt * 16 + low) * DIM;
    const __bf16* brow = wT + (size_t)(nt * 16 + low) * DIM;

    v8f acc = {0.f, 0.f, 0.f, 0.f, 0.f, 0.f, 0.f, 0.f};
#pragma unroll 4
    for (int k0 = 0; k0 < DIM; k0 += 32) {
        v16bf a = load_frag(arow + k0 + base, arow + k0 + 16 + base);
        v16bf b = load_frag(brow + k0 + base, brow + k0 + 16 + base);
        acc = wmma_bf16(a, b, acc);
    }

    const int col = nt * 16 + low;
    const float bv = bias[col];
#pragma unroll
    for (int r = 0; r < 8; r++) {
        int mrow = mt * 16 + r + hi * 8;
        out[(size_t)mrow * DIM + col] = acc[r] + bv;
    }
}

// -------------------------------------------------------------------------
extern "C" void kernel_launch(void* const* d_in, const int* in_sizes, int n_in,
                              void* d_out, int out_size, void* d_ws, size_t ws_size,
                              hipStream_t stream) {
    const float* x        = (const float*)d_in[0];
    const float* ln_gamma = (const float*)d_in[1];
    const float* ln_beta  = (const float*)d_in[2];
    const float* w_qkv    = (const float*)d_in[3];
    const float* b_qkv    = (const float*)d_in[4];
    const float* w_out    = (const float*)d_in[5];
    const float* b_out    = (const float*)d_in[6];
    float* out = (float*)d_out;

    // Workspace layout (bf16 buffers)
    char* ws = (char*)d_ws;
    const size_t SZ_ACT = (size_t)ROWS * DIM * sizeof(__bf16);      // 8 MB
    __bf16* xn     = (__bf16*)(ws);                                  ws += SZ_ACT;
    __bf16* wqkvT  = (__bf16*)(ws);  ws += (size_t)3 * DIM * DIM * sizeof(__bf16);
    __bf16* woutT  = (__bf16*)(ws);  ws += (size_t)DIM * DIM * sizeof(__bf16);
    __bf16* qb     = (__bf16*)(ws);  ws += SZ_ACT;
    __bf16* kb     = (__bf16*)(ws);  ws += SZ_ACT;
    __bf16* vTb    = (__bf16*)(ws);  ws += SZ_ACT;
    __bf16* aoutb  = (__bf16*)(ws);  ws += SZ_ACT;

    // 1. LayerNorm
    ln_kernel<<<ROWS, 256, 0, stream>>>(x, ln_gamma, ln_beta, xn);

    // 2. Weight transpose + bf16 convert
    {
        int e1 = DIM * 3 * DIM;
        transpose_cvt<<<(e1 + 255) / 256, 256, 0, stream>>>(w_qkv, wqkvT, DIM, 3 * DIM);
        int e2 = DIM * DIM;
        transpose_cvt<<<(e2 + 255) / 256, 256, 0, stream>>>(w_out, woutT, DIM, DIM);
    }

    // 3. QKV projection (512 * 96 tiles, 8 tiles/block)
    qkv_gemm<<<(512 * 96) / 8, 256, 0, stream>>>(xn, wqkvT, b_qkv, qb, kb, vTb);

    // 4. Flash attention (8 bh * 64 query tiles)
    attn_kernel<<<8 * 64, 128, 0, stream>>>(qb, kb, vTb, aoutb);

    // 5. Output projection (512 * 32 tiles, 8 tiles/block)
    out_gemm<<<(512 * 32) / 8, 256, 0, stream>>>(aoutb, woutT, b_out, out);
}
